// HybridQLSTM_65481071397548
// MI455X (gfx1250) — matrix operand order repaired and checked by reading, compiled, not run on hardware
//
#include <hip/hip_runtime.h>

typedef __attribute__((ext_vector_type(16))) _Float16 v16h;
typedef __attribute__((ext_vector_type(8)))  float    v8f;

#define T_STEPS 512
#define BATCH   512
#define IN_DIM  64
#define HID     128
#define DTOT    192                 // IN_DIM + HID
#define NROWS   (T_STEPS * BATCH)   // 262144 flattened (t,b) rows
#define NJ      8                   // 4 gates * 2 wires

// ---------------------------------------------------------------------------
// Kernel 1: zx[t*B+b, j] = x[t,b,:64] . Wlin[j,:64] + blin[j]   (j = g*2+w)
// One wave per 16-row tile; two v_wmma_f32_16x16x32_f16 (K=64).
// A layout (16-bit A 16x32, ISA 7.12.2): lane l -> M = l&15; VGPR v holds
//   K = chunk*32 + (v/4)*16 + ((l>=16)?8:0) + 2*(v&3) + {0,1}.
// B layout (16-bit B 32x16): element e of lane l -> K = chunk*32 +
//   (e/8)*16 + ((l>=16)?8:0) + (e&7), N = l&15.  Columns N>=8 zero-padded (branchless).
// C/D layout: VGPR r, lane l -> M = r + 8*(l/16), N = l&15.
// ---------------------------------------------------------------------------
__global__ void __launch_bounds__(256)
zx_wmma_kernel(const float* __restrict__ x,
               const float* __restrict__ Wlin,
               const float* __restrict__ blin,
               float* __restrict__ zx) {
    const int  wave = (blockIdx.x * blockDim.x + threadIdx.x) >> 5;
    const int  lane = threadIdx.x & 31;
    const int  lg   = lane >> 4;      // lane group (0 or 1)
    const int  ln   = lane & 15;
    const long row0 = (long)wave * 16;

    // Branchless lane mask: all-ones for N < 8, zero otherwise.
    const unsigned vmask = (ln < 8) ? 0xFFFFFFFFu : 0u;

    // Fold the bias into the accumulator: C[r] = blin[N] for N<8, else 0.
    // Load from a clamped (always-valid) address, mask the value.
    float bias = __uint_as_float(__float_as_uint(blin[ln & 7]) & vmask);
    v8f c;
#pragma unroll
    for (int r = 0; r < 8; ++r) c[r] = bias;

    const float* xr = x + (row0 + ln) * IN_DIM;          // A row (M = ln)
    const float* wr = Wlin + (ln & 7) * DTOT;            // clamped B row

#pragma unroll
    for (int ch = 0; ch < 2; ++ch) {                     // K chunks 0..31, 32..63
        v16h a, b;
#pragma unroll
        for (int v = 0; v < 8; ++v) {
            int grp = v >> 2, w = v & 3;
            int k0  = ch * 32 + grp * 16 + (lg ? 8 : 0) + 2 * w;
            a[2 * v + 0] = (_Float16)xr[k0];
            a[2 * v + 1] = (_Float16)xr[k0 + 1];
        }
#pragma unroll
        for (int e = 0; e < 16; ++e) {
            int   k  = ch * 32 + (e >> 3) * 16 + (lg ? 8 : 0) + (e & 7);
            // Unconditional load + bitwise zeroing: no EXEC branching.
            float wv = __uint_as_float(__float_as_uint(wr[k]) & vmask);
            b[e] = (_Float16)wv;                          // B[k][n] = W[n][k]
        }
        c = __builtin_amdgcn_wmma_f32_16x16x32_f16(
                /*neg_a=*/false, a, /*neg_b=*/false, b,
                /*c_mod=*/(short)0, c, /*reuse_a=*/false, /*reuse_b=*/false);
    }

    if (ln < 8) {
#pragma unroll
        for (int r = 0; r < 8; ++r) {
            long m = row0 + r + 8 * lg;
            zx[m * NJ + ln] = c[r];
        }
    }
}

// ---------------------------------------------------------------------------
// Fast transcendentals on the hardware v_exp_f32 / v_rcp_f32 paths.
// ---------------------------------------------------------------------------
__device__ __forceinline__ float fast_tanh(float x) {
    // tanh(x) = 1 - 2 / (2^(2*log2(e)*x) + 1)
    float t = __builtin_amdgcn_exp2f(x * 2.885390081777927f);
    return 1.0f - 2.0f * __builtin_amdgcn_rcpf(t + 1.0f);
}
__device__ __forceinline__ float fast_sigmoid(float x) {
    return __builtin_amdgcn_rcpf(1.0f + __builtin_amdgcn_exp2f(x * -1.4426950408889634f));
}

// ---------------------------------------------------------------------------
// Kernel 2: scalar scan. 4 lanes per batch element (one per gate); the four
// gate pre-activations are exchanged with wave shuffles each step.
// Exploits the rank-1 identity: h[b,:] == hbar[b] replicated, so the
// recurrent matvec is hbar * sum(Wlin[j, 64:192]).
// ---------------------------------------------------------------------------
__global__ void __launch_bounds__(256)
scan_kernel(const float* __restrict__ zx,
            const float* __restrict__ Wlin,
            const float* __restrict__ W1, const float* __restrict__ b1,
            const float* __restrict__ W2, const float* __restrict__ b2,
            const float* __restrict__ W3, const float* __restrict__ b3,
            float* __restrict__ hbar_out, float* __restrict__ cfinal) {
    const int gid  = blockIdx.x * blockDim.x + threadIdx.x;  // 0..2047
    const int b    = gid >> 2;
    const int g    = gid & 3;
    const int lane = threadIdx.x & 31;
    const int base = lane & ~3;

    // Recurrent weight row-sums for this gate's two wires.
    float wh0 = 0.0f, wh1 = 0.0f;
    const float* w0p = Wlin + (2 * g + 0) * DTOT + IN_DIM;
    const float* w1p = Wlin + (2 * g + 1) * DTOT + IN_DIM;
    for (int d = 0; d < HID; ++d) { wh0 += w0p[d]; wh1 += w1p[d]; }

    // This gate's MLP weights in registers (broadcast global loads, cached).
    float w1g[8][2], b1g[8], w2g[4][8], b2g[4], w3g[4], b3g;
#pragma unroll
    for (int k = 0; k < 8; ++k) {
        w1g[k][0] = W1[g * 16 + k * 2 + 0];
        w1g[k][1] = W1[g * 16 + k * 2 + 1];
        b1g[k]    = b1[g * 8 + k];
    }
#pragma unroll
    for (int k = 0; k < 4; ++k) {
#pragma unroll
        for (int j = 0; j < 8; ++j) w2g[k][j] = W2[g * 32 + k * 8 + j];
        b2g[k] = b2[g * 4 + k];
        w3g[k] = W3[g * 4 + k];
    }
    b3g = b3[g];

    float hbar = 0.0f, cbar = 0.0f;
    for (int t = 0; t < T_STEPS; ++t) {
        const float* zp = zx + ((long)t * BATCH + b) * NJ + 2 * g;
        float z0 = zp[0] + hbar * wh0;   // blin already folded into zx
        float z1 = zp[1] + hbar * wh1;

        float h1v[8];
#pragma unroll
        for (int k = 0; k < 8; ++k)
            h1v[k] = fast_tanh(w1g[k][0] * z0 + w1g[k][1] * z1 + b1g[k]);

        float h2v[4];
#pragma unroll
        for (int k = 0; k < 4; ++k) {
            float s = b2g[k];
#pragma unroll
            for (int j = 0; j < 8; ++j) s += w2g[k][j] * h1v[j];
            h2v[k] = fast_tanh(s);
        }

        float zz = b3g;
#pragma unroll
        for (int k = 0; k < 4; ++k) zz += w3g[k] * h2v[k];

        // Exchange the four gate pre-activations within the lane quad.
        float zf = __shfl(zz, base + 0, 32);
        float zi = __shfl(zz, base + 1, 32);
        float zu = __shfl(zz, base + 2, 32);
        float zo = __shfl(zz, base + 3, 32);

        float f  = fast_sigmoid(zf);
        float ii = fast_sigmoid(zi);
        float gg = fast_tanh(fast_sigmoid(zu));
        float oo = fast_sigmoid(zo);

        cbar = f * cbar + ii * gg;
        hbar = oo * fast_tanh(cbar);

        if (g == 0) hbar_out[t * BATCH + b] = hbar;
    }
    if (g == 0) cfinal[b] = cbar;
}

// ---------------------------------------------------------------------------
// Kernel 3: broadcast hbar/cfinal into the full [T,B,128] ‖ [B,128] ‖ [B,128]
// output with float4 stores (bandwidth-bound, ~135 MB).
// ---------------------------------------------------------------------------
__global__ void __launch_bounds__(256)
broadcast_kernel(const float* __restrict__ hbar,
                 const float* __restrict__ cfinal,
                 float4* __restrict__ out) {
    const long S4 = (long)NROWS * (HID / 4);   // stacked float4 count
    const long H4 = (long)BATCH * (HID / 4);
    long i = (long)blockIdx.x * blockDim.x + threadIdx.x;
    if (i >= S4 + 2 * H4) return;
    float v;
    if (i < S4)              v = hbar[i >> 5];                                   // HID/4 == 32
    else if (i < S4 + H4)    v = hbar[(long)(T_STEPS - 1) * BATCH + ((i - S4) >> 5)];
    else                     v = cfinal[(i - S4 - H4) >> 5];
    out[i] = make_float4(v, v, v, v);
}

extern "C" void kernel_launch(void* const* d_in, const int* in_sizes, int n_in,
                              void* d_out, int out_size, void* d_ws, size_t ws_size,
                              hipStream_t stream) {
    const float* x    = (const float*)d_in[0];
    const float* Wlin = (const float*)d_in[1];
    const float* blin = (const float*)d_in[2];
    const float* W1   = (const float*)d_in[3];
    const float* b1   = (const float*)d_in[4];
    const float* W2   = (const float*)d_in[5];
    const float* b2   = (const float*)d_in[6];
    const float* W3   = (const float*)d_in[7];
    const float* b3   = (const float*)d_in[8];
    float* out = (float*)d_out;

    float* zx     = (float*)d_ws;                 // NROWS * 8 floats (8 MB)
    float* hbar   = zx + (long)NROWS * NJ;        // NROWS floats (1 MB)
    float* cfinal = hbar + NROWS;                 // BATCH floats

    // K1: 16384 waves (one 16-row WMMA tile each) = 2048 blocks x 256 threads
    zx_wmma_kernel<<<2048, 256, 0, stream>>>(x, Wlin, blin, zx);

    // K2: 2048 threads = 4 lanes per batch element
    scan_kernel<<<8, 256, 0, stream>>>(zx, Wlin, W1, b1, W2, b2, W3, b3,
                                       hbar, cfinal);

    // K3: broadcast to full output
    long total4 = (long)NROWS * (HID / 4) + 2L * BATCH * (HID / 4);
    int  blocks = (int)((total4 + 255) / 256);
    broadcast_kernel<<<blocks, 256, 0, stream>>>(hbar, cfinal, (float4*)out);
}
